// SSIM_53498112639400
// MI455X (gfx1250) — compile-verified, matrix operand order checked by reading, not run on hardware
//
#include <hip/hip_runtime.h>

typedef __attribute__((ext_vector_type(16))) _Float16 v16h;
typedef __attribute__((ext_vector_type(8)))  float    v8f;

#define IMG_H   512
#define IMG_W   512
#define PLANES  48            // 16 batch * 3 channels
#define TILE_R  16
#define TILE_C  64
#define PATCH_R 26            // 16 + 10 halo
#define PATCH_C 74            // 64 + 10 halo
#define PAD_R   32            // K-dim padding for 16x16x32 WMMA
#define PAD_C   80

__global__ void ssim_zero_kernel(float* out) { out[0] = 0.0f; }

// K index held by vector element e for both the f16 A(16x32) and B(32x16) operands.
__device__ __forceinline__ int kmap(int e, int hh) {
    return e + 8 * hh + ((e >= 8) ? 8 : 0);
}

__device__ __forceinline__ unsigned pack2h(float a, float b) {
    union { _Float16 h[2]; unsigned u; } c;
    c.h[0] = (_Float16)a;
    c.h[1] = (_Float16)b;
    return c.u;
}

__global__ void __launch_bounds__(32)
ssim_wmma_kernel(const float* __restrict__ pred,
                 const float* __restrict__ targ,
                 const float* __restrict__ k2,   // (3,1,11,11), all channels identical
                 float* __restrict__ out)
{
    __shared__ float    rawP[PATCH_R][PAD_C];    // 8320 B, async-staged fp32
    __shared__ float    rawT[PATCH_R][PAD_C];    // 8320 B
    __shared__ _Float16 patch[5][PAD_R][PAD_C];  // 25600 B  (p, t, pp, tt, pt)
    __shared__ _Float16 mid[5][TILE_R][PAD_C];   // 12800 B

    const int lane = threadIdx.x;
    const int p16  = lane & 15;   // row index (A) / col index (B,C,D)
    const int hh   = lane >> 4;   // half-wave select

    // tile decode: 48 planes * (32 row-tiles) * (8 col-tiles) = 12288 blocks
    const int bid   = blockIdx.x;
    const int plane = bid >> 8;
    const int rem   = bid & 255;
    const int tr    = rem >> 3;
    const int tc    = rem & 7;
    const int r0    = tr * TILE_R;
    const int c0    = tc * TILE_C;

    const float* __restrict__ P = pred + (size_t)plane * IMG_H * IMG_W;
    const float* __restrict__ T = targ + (size_t)plane * IMG_H * IMG_W;

    // ---- fire off async global->LDS staging of the raw fp32 patches ----
    // GVS mode: SGPR base + 32-bit VGPR byte offset. Only in-bounds lanes
    // issue (EXEC mask); out-of-bounds LDS slots are never consumed.
    for (int row = 0; row < PATCH_R; ++row) {
        int ir = r0 - 5 + row;
        #pragma unroll
        for (int ch = 0; ch < 3; ++ch) {
            int col = ch * 32 + lane;
            int ic  = c0 - 5 + col;
            if (col < PATCH_C && (unsigned)ir < (unsigned)IMG_H &&
                (unsigned)ic < (unsigned)IMG_W) {
                unsigned lp = (unsigned)(size_t)&rawP[row][col];  // LDS byte offset
                unsigned lt = (unsigned)(size_t)&rawT[row][col];
                int go = (ir * IMG_W + ic) * 4;
                asm volatile("global_load_async_to_lds_b32 %0, %1, %2"
                             :: "v"(lp), "v"(go), "s"(P) : "memory");
                asm volatile("global_load_async_to_lds_b32 %0, %1, %2"
                             :: "v"(lt), "v"(go), "s"(T) : "memory");
            }
        }
    }

    // ---- overlap: build weight operand while the async copies fly ----
    // 1-D normalized gaussian from the 2-D kernel input:
    // k2[5][j] = g[5]*g[j], g[5] = sqrt(k2[5][5])  ->  g[j] = k2[5][j]/sqrt(k2[5][5])
    const float gc = sqrtf(k2[5 * 11 + 5]);
    float G[11];
    #pragma unroll
    for (int i = 0; i < 11; ++i) G[i] = k2[5 * 11 + i] / gc;

    // Banded weight operand: element e = g[K(e) - p16] (0 outside band).
    // Serves as A for the vertical pass (Wv[r,j]=g[j-r]) and as B for the
    // horizontal pass (Wh[j,c]=g[j-c]): identical per-lane contents.
    v16h wgt;
    #pragma unroll
    for (int e = 0; e < 16; ++e) {
        int j = kmap(e, hh);
        int d = j - p16;
        wgt[e] = (d >= 0 && d <= 10) ? (_Float16)G[d] : (_Float16)0.0f;
    }

    // ---- retire the async copies, then build the 5 f16 quantity patches ----
    asm volatile("s_wait_asynccnt 0" ::: "memory");
    __syncthreads();

    for (int row = 0; row < PAD_R; ++row) {
        int ir = r0 - 5 + row;
        #pragma unroll
        for (int ch = 0; ch < 2; ++ch) {
            int pc = ch * 32 + lane;          // column-pair index
            if (pc >= PAD_C / 2) continue;    // 40 pairs per row
            int col0 = 2 * pc, col1 = col0 + 1;
            float a0 = 0.0f, a1 = 0.0f, b0 = 0.0f, b1 = 0.0f;
            if (row < PATCH_R) {
                float2 vp = *(const float2*)&rawP[row][col0];
                float2 vt = *(const float2*)&rawT[row][col0];
                int ic0 = c0 - 5 + col0, ic1 = ic0 + 1;
                bool ok0 = (col0 < PATCH_C) && ((unsigned)ir < (unsigned)IMG_H) &&
                           ((unsigned)ic0 < (unsigned)IMG_W);
                bool ok1 = (col1 < PATCH_C) && ((unsigned)ir < (unsigned)IMG_H) &&
                           ((unsigned)ic1 < (unsigned)IMG_W);
                a0 = ok0 ? vp.x : 0.0f;  a1 = ok1 ? vp.y : 0.0f;
                b0 = ok0 ? vt.x : 0.0f;  b1 = ok1 ? vt.y : 0.0f;
            }
            *(unsigned*)&patch[0][row][col0] = pack2h(a0, a1);
            *(unsigned*)&patch[1][row][col0] = pack2h(b0, b1);
            *(unsigned*)&patch[2][row][col0] = pack2h(a0 * a0, a1 * a1);
            *(unsigned*)&patch[3][row][col0] = pack2h(b0 * b0, b1 * b1);
            *(unsigned*)&patch[4][row][col0] = pack2h(a0 * b0, a1 * b1);
        }
    }
    __syncthreads();

    // ---- vertical pass: mid[q](16x74) = Wv(16x26) x patch[q](26x74) ----
    for (int q = 0; q < 5; ++q) {
        for (int ct = 0; ct < 5; ++ct) {          // 5 column tiles of 16
            v16h b;
            #pragma unroll
            for (int e = 0; e < 16; ++e) {
                int j = kmap(e, hh);              // K = patch row (rows >=26 zero)
                b[e] = patch[q][j][ct * 16 + p16];
            }
            v8f acc = {};
            acc = __builtin_amdgcn_wmma_f32_16x16x32_f16(
                      false, wgt, false, b, (short)0, acc, false, false);
            #pragma unroll
            for (int r = 0; r < 8; ++r) {
                int m   = r + 8 * hh;             // C/D layout: M = vgpr + 8*(lane>>4)
                int col = ct * 16 + p16;
                float v = (col < PATCH_C) ? acc[r] : 0.0f;  // keep pad columns zero
                mid[q][m][col] = (_Float16)v;
            }
        }
    }
    __syncthreads();

    // ---- horizontal pass + SSIM map + partial mean ----
    const float C1 = 1e-4f, C2 = 9e-4f;
    float lsum = 0.0f;

    for (int ct = 0; ct < 4; ++ct) {              // 4 output tiles of 16x16
        v8f o[5];
        #pragma unroll
        for (int q = 0; q < 5; ++q) {
            v16h a;
            #pragma unroll
            for (int e = 0; e < 16; ++e) {
                int j = kmap(e, hh);              // K = mid column offset
                a[e] = mid[q][p16][ct * 16 + j];
            }
            v8f acc = {};
            o[q] = __builtin_amdgcn_wmma_f32_16x16x32_f16(
                       false, a, false, wgt, (short)0, acc, false, false);
        }
        #pragma unroll
        for (int r = 0; r < 8; ++r) {
            float mu1 = o[0][r], mu2 = o[1][r];
            float xx  = o[2][r], yy  = o[3][r], xy = o[4][r];
            float m11 = mu1 * mu1, m22 = mu2 * mu2, m12 = mu1 * mu2;
            float s1  = xx - m11, s2 = yy - m22, s12 = xy - m12;
            float num = (2.0f * m12 + C1) * (2.0f * s12 + C2);
            float den = (m11 + m22 + C1) * (s1 + s2 + C2);
            lsum += num / den;
        }
    }

    // wave32 reduction, then accumulate mean contribution
    #pragma unroll
    for (int off = 16; off > 0; off >>= 1)
        lsum += __shfl_down(lsum, off, 32);
    if (lane == 0) {
        const float invN = 1.0f / (float)((size_t)PLANES * IMG_H * IMG_W);
        atomicAdd(out, lsum * invN);
    }
}

extern "C" void kernel_launch(void* const* d_in, const int* in_sizes, int n_in,
                              void* d_out, int out_size, void* d_ws, size_t ws_size,
                              hipStream_t stream) {
    (void)in_sizes; (void)n_in; (void)out_size; (void)d_ws; (void)ws_size;
    const float* pred = (const float*)d_in[0];
    const float* targ = (const float*)d_in[1];
    const float* kern = (const float*)d_in[2];
    float* out = (float*)d_out;

    ssim_zero_kernel<<<1, 1, 0, stream>>>(out);

    const int blocks = PLANES * (IMG_H / TILE_R) * (IMG_W / TILE_C);  // 12288
    ssim_wmma_kernel<<<blocks, 32, 0, stream>>>(pred, targ, kern, out);
}
